// SwinTransformerBlock_27822798143647
// MI455X (gfx1250) — matrix operand order repaired and checked by reading, not compiled, not run on hardware
//
#include <hip/hip_runtime.h>
#include <hip/hip_bf16.h>

// ---------------------------------------------------------------------------
// Swin Transformer block for MI455X (gfx1250), wave32.
// WMMA f32_16x16x32_f16 + async global->LDS staging (ASYNCcnt pipeline).
// ---------------------------------------------------------------------------

typedef __attribute__((ext_vector_type(16))) _Float16 v16h;
typedef __attribute__((ext_vector_type(8)))  float    v8f;
typedef __attribute__((ext_vector_type(4)))  int      vi4;

#define AS1 __attribute__((address_space(1)))
#define AS3 __attribute__((address_space(3)))

// 16-byte global -> LDS async copy (per-lane), tracked by ASYNCcnt.
#if __has_builtin(__builtin_amdgcn_global_load_async_to_lds_b128)
__device__ __forceinline__ void async_cp16(_Float16* lds, const _Float16* g) {
  __builtin_amdgcn_global_load_async_to_lds_b128(
      (AS1 vi4*)g, (AS3 vi4*)lds, 0, 0);
}
#else
__device__ __forceinline__ void async_cp16(_Float16* lds, const _Float16* g) {
  unsigned lo = (unsigned)(size_t)lds;                 // LDS byte offset
  unsigned long long ga = (unsigned long long)(size_t)g;
  asm volatile("global_load_async_to_lds_b128 %0, %1, off"
               :: "v"(lo), "v"(ga) : "memory");
}
#endif

__device__ __forceinline__ void wait_async0() {
#if __has_builtin(__builtin_amdgcn_s_wait_asynccnt)
  __builtin_amdgcn_s_wait_asynccnt(0);
#else
  asm volatile("s_wait_asynccnt 0x0" ::: "memory");
#endif
}

union V16U { v16h v; uint4 q[2]; };

// A-matrix fragment (16x32 f16): lane l holds row m=l&15; halves e:
//   k = (e/8)*16 + (l>>4)*8 + (e%8)  -> two 16B chunks at p and p+16
__device__ __forceinline__ v16h frag_a(const _Float16* p) {
  V16U t;
  t.q[0] = *reinterpret_cast<const uint4*>(p);
  t.q[1] = *reinterpret_cast<const uint4*>(p + 16);
  return t.v;
}
// B-matrix fragment (32x16 f16, stored col-major [n][k]): lane l holds col
// n=l&15; halves e: k = (e/8)*16 + (l>>4)*8 + (e%8) -> two 16B chunks at p and p+16
__device__ __forceinline__ v16h frag_b(const _Float16* p) {
  V16U t;
  t.q[0] = *reinterpret_cast<const uint4*>(p);
  t.q[1] = *reinterpret_cast<const uint4*>(p + 16);
  return t.v;
}
__device__ __forceinline__ v8f wmma16(v16h a, v16h b, v8f c) {
  return __builtin_amdgcn_wmma_f32_16x16x32_f16(false, a, false, b,
                                                (short)0, c, false, false);
}

// ---------------------------------------------------------------------------
// Weight convert + transpose: W[K][N] f32 -> WT[N][K] f16
// ---------------------------------------------------------------------------
__global__ __launch_bounds__(256) void swin_wcvt(const float* __restrict__ W,
                                                 _Float16* __restrict__ WT,
                                                 int K, int N) {
  int idx = blockIdx.x * 256 + threadIdx.x;
  if (idx >= K * N) return;
  int n = idx / K, k = idx % K;
  WT[idx] = (_Float16)W[(size_t)k * N + n];
}

// ---------------------------------------------------------------------------
// Relative-position bias expand: bias_table[169][6] -> biasNH[6][64][64] f32
// (padded cols/rows = -1e30 so softmax over the 49->64 padding yields 0)
// ---------------------------------------------------------------------------
__global__ __launch_bounds__(256) void swin_bias(const float* __restrict__ bt,
                                                 float* __restrict__ biasNH) {
  int idx = blockIdx.x * 256 + threadIdx.x;
  if (idx >= 6 * 64 * 64) return;
  int h = idx >> 12, i = (idx >> 6) & 63, j = idx & 63;
  float v = -1e30f;
  if (i < 49 && j < 49) {
    int dh = (i / 7 - j / 7) + 6;
    int dw = (i % 7 - j % 7) + 6;
    v = bt[(dh * 13 + dw) * 6 + h];
  }
  biasNH[idx] = v;
}

// ---------------------------------------------------------------------------
// LayerNorm (+ optional cyclic shift & window partition) -> f16 token matrix.
// One wave per token; lane handles 6 channels (192 = 32*6).
// ---------------------------------------------------------------------------
__global__ __launch_bounds__(256) void swin_ln(const float* __restrict__ xin,
                                               const float* __restrict__ g,
                                               const float* __restrict__ be,
                                               _Float16* __restrict__ outp,
                                               int windowed) {
  int t = blockIdx.x * 8 + (threadIdx.x >> 5);
  int lane = threadIdx.x & 31;
  size_t src;
  if (windowed) {
    int w = t / 49, nt = t % 49;
    int b = w >> 6, wh = (w >> 3) & 7, ww = w & 7;
    int r7 = nt / 7, c7 = nt % 7;
    int hh = (wh * 7 + r7 + 3) % 56;   // SHIFT = 3
    int wp = (ww * 7 + c7 + 3) % 56;
    src = ((size_t)b * 56 + hh) * 56 + wp;
  } else {
    src = (size_t)t;
  }
  const float* px = xin + src * 192;
  float v[6];
  float s = 0.f;
#pragma unroll
  for (int i = 0; i < 6; ++i) { v[i] = px[lane + 32 * i]; s += v[i]; }
#pragma unroll
  for (int o = 16; o >= 1; o >>= 1) s += __shfl_xor(s, o, 32);
  float mean = s * (1.f / 192.f);
  float s2 = 0.f;
#pragma unroll
  for (int i = 0; i < 6; ++i) { float d = v[i] - mean; s2 += d * d; }
#pragma unroll
  for (int o = 16; o >= 1; o >>= 1) s2 += __shfl_xor(s2, o, 32);
  float inv = rsqrtf(s2 * (1.f / 192.f) + 1e-5f);
#pragma unroll
  for (int i = 0; i < 6; ++i) {
    int ch = lane + 32 * i;
    outp[(size_t)t * 192 + ch] = (_Float16)((v[i] - mean) * inv * g[ch] + be[ch]);
  }
}

// ---------------------------------------------------------------------------
// Tiled WMMA GEMM: C[M x N] = A[M x K] * BT[N x K]^T, f16 in, f32 accum.
// WG tile 128x64, 8 waves (4x2), wave tile 32x32 = 2x2 WMMA, K step 32.
// Double-buffered LDS, async global->LDS staging overlapped with WMMA.
// EPI: 0 = QKV split, 1 = proj + residual + reverse shift, 2 = bias+GELU,
//      3 = bias + residual -> output
// ---------------------------------------------------------------------------
#define LDA 40   // 32 + 8 halves pad -> rows stay 16B aligned, few conflicts

template <int EPI>
__global__ __launch_bounds__(256) void swin_gemm(
    const _Float16* __restrict__ A, const _Float16* __restrict__ BT,
    int K, int N,
    const float* __restrict__ cbias, const float* __restrict__ res,
    float* __restrict__ of32,
    _Float16* __restrict__ oh0, _Float16* __restrict__ oh1,
    _Float16* __restrict__ oh2) {
  __shared__ _Float16 Al[2][128 * LDA];
  __shared__ _Float16 Bl[2][64 * LDA];

  const int m0 = blockIdx.x * 128;
  const int n0 = blockIdx.y * 64;
  const int tid = threadIdx.x;
  const int lane = tid & 31, wv = tid >> 5;
  const int wm = wv & 3, wn = wv >> 2;          // 4 x 2 wave grid
  const int hi = lane >> 4, lo = lane & 15;

  // per-thread staging coordinates (16B chunks)
  const int ar0 = tid >> 2, ac0 = (tid & 3) * 8;            // A chunk 0
  const int ar1 = (tid + 256) >> 2, ac1 = ac0;              // A chunk 1
  const int br  = tid >> 2, bc = (tid & 3) * 8;             // B chunk

  auto stage = [&](int buf, int k0) {
    async_cp16(&Al[buf][ar0 * LDA + ac0], &A[(size_t)(m0 + ar0) * K + k0 + ac0]);
    async_cp16(&Al[buf][ar1 * LDA + ac1], &A[(size_t)(m0 + ar1) * K + k0 + ac1]);
    async_cp16(&Bl[buf][br * LDA + bc],   &BT[(size_t)(n0 + br) * K + k0 + bc]);
  };

  v8f acc[2][2];
  const v8f z8 = {0.f, 0.f, 0.f, 0.f, 0.f, 0.f, 0.f, 0.f};
#pragma unroll
  for (int t = 0; t < 2; ++t)
#pragma unroll
    for (int u = 0; u < 2; ++u) acc[t][u] = z8;

  stage(0, 0);
  wait_async0();
  __syncthreads();

  int buf = 0;
  for (int k0 = 0; k0 < K; k0 += 32) {
    if (k0 + 32 < K) stage(buf ^ 1, k0 + 32);   // prefetch next chunk (async)

    v16h af[2], bf[2];
#pragma unroll
    for (int t = 0; t < 2; ++t)
      af[t] = frag_a(&Al[buf][(wm * 32 + t * 16 + lo) * LDA + hi * 8]);
#pragma unroll
    for (int u = 0; u < 2; ++u)
      bf[u] = frag_b(&Bl[buf][(wn * 32 + u * 16 + lo) * LDA + hi * 8]);
#pragma unroll
    for (int t = 0; t < 2; ++t)
#pragma unroll
      for (int u = 0; u < 2; ++u) acc[t][u] = wmma16(af[t], bf[u], acc[t][u]);

    wait_async0();       // my async writes for buf^1 are done
    __syncthreads();     // everyone done writing buf^1 / reading buf
    buf ^= 1;
  }

  // epilogue: lane (l) reg (r) of acc[t][u] holds (m, n) below
#pragma unroll
  for (int t = 0; t < 2; ++t) {
#pragma unroll
    for (int u = 0; u < 2; ++u) {
#pragma unroll
      for (int r = 0; r < 8; ++r) {
        int m = m0 + wm * 32 + t * 16 + r + hi * 8;
        int n = n0 + wn * 32 + u * 16 + lo;
        float val = acc[t][u][r];
        if constexpr (EPI == 0) {                       // QKV split
          val += cbias[n];
          int qi = n / 192, hh = (n % 192) >> 5, d = n & 31;
          if (qi == 0) val *= 0.17677669529663687f;     // 1/sqrt(32)
          size_t w = (size_t)(m / 49), nt = (size_t)(m % 49);
          size_t off = ((w * 6 + hh) * 49 + nt) * 32 + d;
          _Float16* dst = (qi == 0) ? oh0 : (qi == 1) ? oh1 : oh2;
          dst[off] = (_Float16)val;
        } else if constexpr (EPI == 1) {                // proj + residual
          int w = m / 49, nt = m % 49;
          int b = w >> 6, wh = (w >> 3) & 7, ww = w & 7;
          int r7 = nt / 7, c7 = nt % 7;
          int hh = (wh * 7 + r7 + 3) % 56;
          int wp = (ww * 7 + c7 + 3) % 56;
          size_t pix = ((size_t)b * 56 + hh) * 56 + wp;
          float o = val + cbias[n] + res[pix * 192 + n];
          of32[pix * 192 + n] = o;
        } else if constexpr (EPI == 2) {                // bias + exact GELU
          float a = val + cbias[n];
          float ge = 0.5f * a * (1.0f + erff(a * 0.70710678118654752f));
          oh0[(size_t)m * 768 + n] = (_Float16)ge;
        } else {                                        // bias + residual out
          of32[(size_t)m * 192 + n] = val + cbias[n] + res[(size_t)m * 192 + n];
        }
      }
    }
  }
}

// ---------------------------------------------------------------------------
// Window attention: one block per (window, head). 128 threads = 4 waves.
// S = Q K^T (WMMA) + bias, softmax, O = P V (WMMA).
// Q/K tiles staged with async LDS copies; V transposed through VGPRs.
// ---------------------------------------------------------------------------
#define LDV 72   // 64 + 8 halves pad

__global__ __launch_bounds__(128) void swin_attn(
    const _Float16* __restrict__ Qb, const _Float16* __restrict__ Kb,
    const _Float16* __restrict__ Vb, const float* __restrict__ biasNH,
    _Float16* __restrict__ aout) {
  __shared__ _Float16 Qs[64 * LDA];
  __shared__ _Float16 Ks[64 * LDA];
  __shared__ _Float16 Vt[32 * LDV];   // transposed: [d][key]
  __shared__ _Float16 Ps[64 * LDV];

  const int wh = blockIdx.x;           // w*6 + h
  const int w = wh / 6, h = wh % 6;
  const int tid = threadIdx.x;
  const int lane = tid & 31, wv = tid >> 5;
  const int hi = lane >> 4, lo = lane & 15;
  const size_t base = (size_t)wh * 49 * 32;

  // load Q, K (64x32, zero padded) and V transposed (32 x 64)
  {
    int row = tid >> 1;
    int cp = (tid & 1) * 16;
    uint4 z4 = {0u, 0u, 0u, 0u};
    if (row < 49) {
      async_cp16(&Qs[row * LDA + cp],     &Qb[base + row * 32 + cp]);
      async_cp16(&Qs[row * LDA + cp + 8], &Qb[base + row * 32 + cp + 8]);
      async_cp16(&Ks[row * LDA + cp],     &Kb[base + row * 32 + cp]);
      async_cp16(&Ks[row * LDA + cp + 8], &Kb[base + row * 32 + cp + 8]);
      V16U vt;
      vt.q[0] = *reinterpret_cast<const uint4*>(&Vb[base + row * 32 + cp]);
      vt.q[1] = *reinterpret_cast<const uint4*>(&Vb[base + row * 32 + cp + 8]);
#pragma unroll
      for (int j = 0; j < 16; ++j) Vt[(cp + j) * LDV + row] = vt.v[j];
    } else {
      *reinterpret_cast<uint4*>(&Qs[row * LDA + cp]) = z4;
      *reinterpret_cast<uint4*>(&Qs[row * LDA + cp + 8]) = z4;
      *reinterpret_cast<uint4*>(&Ks[row * LDA + cp]) = z4;
      *reinterpret_cast<uint4*>(&Ks[row * LDA + cp + 8]) = z4;
#pragma unroll
      for (int j = 0; j < 16; ++j) Vt[(cp + j) * LDV + row] = (_Float16)0.f;
    }
  }
  wait_async0();
  __syncthreads();

  const int m0 = wv * 16;   // this wave's query strip
  const v8f z8 = {0.f, 0.f, 0.f, 0.f, 0.f, 0.f, 0.f, 0.f};

  // S = Q K^T   (16x64 per wave, K-dim = 32 -> one WMMA per n-tile)
  v16h aq = frag_a(&Qs[(m0 + lo) * LDA + hi * 8]);
  v8f sacc[4];
#pragma unroll
  for (int u = 0; u < 4; ++u) {
    v16h bk = frag_b(&Ks[(u * 16 + lo) * LDA + hi * 8]);
    sacc[u] = wmma16(aq, bk, z8);
  }

  // bias + softmax over 64 (padded) keys; write P to LDS in f16
  const float* bb = biasNH + (size_t)h * 4096;
#pragma unroll
  for (int r = 0; r < 8; ++r) {
    int i = m0 + r + hi * 8;
    float vv[4];
    float mx = -3e38f;
#pragma unroll
    for (int u = 0; u < 4; ++u) {
      vv[u] = sacc[u][r] + bb[i * 64 + u * 16 + lo];
      mx = fmaxf(mx, vv[u]);
    }
#pragma unroll
    for (int o = 1; o < 16; o <<= 1) mx = fmaxf(mx, __shfl_xor(mx, o, 32));
    float sum = 0.f;
#pragma unroll
    for (int u = 0; u < 4; ++u) { vv[u] = __expf(vv[u] - mx); sum += vv[u]; }
#pragma unroll
    for (int o = 1; o < 16; o <<= 1) sum += __shfl_xor(sum, o, 32);
    float rinv = 1.0f / sum;
#pragma unroll
    for (int u = 0; u < 4; ++u)
      Ps[i * LDV + u * 16 + lo] = (_Float16)(vv[u] * rinv);
  }

  // O = P V  (16x32 per wave, K-dim 64 -> two chunks)
  v8f oacc[2] = {z8, z8};
#pragma unroll
  for (int c = 0; c < 2; ++c) {
    v16h af = frag_a(&Ps[(m0 + lo) * LDV + c * 32 + hi * 8]);
#pragma unroll
    for (int u = 0; u < 2; ++u) {
      v16h bf = frag_b(&Vt[(u * 16 + lo) * LDV + c * 32 + hi * 8]);
      oacc[u] = wmma16(af, bf, oacc[u]);
    }
  }
#pragma unroll
  for (int u = 0; u < 2; ++u) {
#pragma unroll
    for (int r = 0; r < 8; ++r) {
      int m = m0 + r + hi * 8;
      if (m < 49) {
        int d = u * 16 + lo;
        aout[((size_t)w * 49 + m) * 192 + h * 32 + d] = (_Float16)oacc[u][r];
      }
    }
  }
}

// ---------------------------------------------------------------------------
extern "C" void kernel_launch(void* const* d_in, const int* in_sizes, int n_in,
                              void* d_out, int out_size, void* d_ws,
                              size_t ws_size, hipStream_t stream) {
  (void)in_sizes; (void)n_in; (void)out_size; (void)ws_size;
  const float* x      = (const float*)d_in[0];
  const float* qkv_w  = (const float*)d_in[1];
  const float* qkv_b  = (const float*)d_in[2];
  const float* proj_w = (const float*)d_in[3];
  const float* proj_b = (const float*)d_in[4];
  const float* btab   = (const float*)d_in[5];
  const float* n1g    = (const float*)d_in[6];
  const float* n1b    = (const float*)d_in[7];
  const float* n2g    = (const float*)d_in[8];
  const float* n2b    = (const float*)d_in[9];
  const float* w1     = (const float*)d_in[10];
  const float* b1     = (const float*)d_in[11];
  const float* w2     = (const float*)d_in[12];
  const float* b2     = (const float*)d_in[13];
  float* out = (float*)d_out;

  const size_t TOK = 50176;   // 1024 windows * 49 tokens
  char* ws = (char*)d_ws;
  size_t off = 0;
  auto take = [&](size_t bytes) {
    size_t o = off;
    off += (bytes + 255) & ~(size_t)255;
    return o;
  };
  _Float16* wt_qkv  = (_Float16*)(ws + take(576 * 192 * 2));
  _Float16* wt_proj = (_Float16*)(ws + take(192 * 192 * 2));
  _Float16* wt_1    = (_Float16*)(ws + take(768 * 192 * 2));
  _Float16* wt_2    = (_Float16*)(ws + take(192 * 768 * 2));
  float*    biasNH  = (float*)   (ws + take(6 * 64 * 64 * 4));
  _Float16* bufA    = (_Float16*)(ws + take(TOK * 192 * 2));       // LN1 out -> attn out -> LN2 out
  size_t o_qkv = take(3 * TOK * 192 * 2);                          // Q,K,V; later aliased by x2
  _Float16* Qb = (_Float16*)(ws + o_qkv);
  _Float16* Kb = Qb + TOK * 192;
  _Float16* Vb = Kb + TOK * 192;
  float*    x2 = (float*)(ws + o_qkv);                             // 38.5MB <= 57.8MB region
  _Float16* mid = (_Float16*)(ws + take(TOK * 768 * 2));

  // 1) weights -> f16, transposed [N][K]
  swin_wcvt<<<(110592 + 255) / 256, 256, 0, stream>>>(qkv_w, wt_qkv, 192, 576);
  swin_wcvt<<<(36864 + 255) / 256, 256, 0, stream>>>(proj_w, wt_proj, 192, 192);
  swin_wcvt<<<(147456 + 255) / 256, 256, 0, stream>>>(w1, wt_1, 192, 768);
  swin_wcvt<<<(147456 + 255) / 256, 256, 0, stream>>>(w2, wt_2, 768, 192);
  // 2) relative-position bias expand
  swin_bias<<<96, 256, 0, stream>>>(btab, biasNH);
  // 3) LN1 + shift + window partition
  swin_ln<<<6272, 256, 0, stream>>>(x, n1g, n1b, bufA, 1);
  // 4) QKV GEMM (M=50176, K=192, N=576)
  swin_gemm<0><<<dim3(392, 9), 256, 0, stream>>>(bufA, wt_qkv, 192, 576,
      qkv_b, nullptr, nullptr, Qb, Kb, Vb);
  // 5) window attention (6144 window-head blocks)
  swin_attn<<<6144, 128, 0, stream>>>(Qb, Kb, Vb, biasNH, bufA);
  // 6) proj GEMM + reverse shift + residual -> x2
  swin_gemm<1><<<dim3(392, 3), 256, 0, stream>>>(bufA, wt_proj, 192, 192,
      proj_b, x, x2, nullptr, nullptr, nullptr);
  // 7) LN2
  swin_ln<<<6272, 256, 0, stream>>>(x2, n2g, n2b, bufA, 0);
  // 8) MLP fc1 + GELU (N=768)
  swin_gemm<2><<<dim3(392, 12), 256, 0, stream>>>(bufA, wt_1, 192, 768,
      b1, nullptr, nullptr, mid, nullptr, nullptr);
  // 9) MLP fc2 + residual -> out
  swin_gemm<3><<<dim3(392, 3), 256, 0, stream>>>(mid, wt_2, 768, 192,
      b2, x2, out, nullptr, nullptr, nullptr);
}
